// PPM_2104533975450
// MI455X (gfx1250) — matrix operand order repaired and checked by reading, not compiled, not run on hardware
//
#include <hip/hip_runtime.h>
#include <hip/hip_bf16.h>
#include <math.h>

#define B_  32
#define C_  256
#define HW_ 1024
#define EPS_ 1e-8f

typedef __attribute__((ext_vector_type(16))) __bf16 bf16x16;
typedef __attribute__((ext_vector_type(8)))  float  f32x8;

union FragU { bf16x16 v; uint4 u[2]; };

// A fragment (16x32 bf16, M x K), source stored row-major [M][K], lda in elems.
// Layout (ISA 7.12.2): lanes 0-15: V0-3 = K0..7, V4-7 = K16..23 (M = lane)
//                      lanes16-31: V0-3 = K8..15, V4-7 = K24..31 (M = lane-16)
__device__ __forceinline__ bf16x16 load_a_frag(const __bf16* A, int lda, int lane) {
  int m  = lane & 15;
  int k0 = (lane < 16) ? 0 : 8;
  FragU f;
  f.u[0] = *(const uint4*)(A + m * lda + k0);
  f.u[1] = *(const uint4*)(A + m * lda + k0 + 16);
  return f.v;
}

// B fragment (32x16 bf16, K x N), source stored as B^T row-major [N][K], ldb in elems.
// Layout: lanes 0-15: V0-3 = K0..7, V4-7 = K16..23 (N = lane); lanes 16-31: V0-3 = K8..15, V4-7 = K24..31.
__device__ __forceinline__ bf16x16 load_b_frag(const __bf16* BT, int ldb, int lane) {
  int n  = lane & 15;
  int k0 = (lane < 16) ? 0 : 8;
  FragU f;
  f.u[0] = *(const uint4*)(BT + n * ldb + k0);
  f.u[1] = *(const uint4*)(BT + n * ldb + k0 + 16);
  return f.v;
}

#define WMMA_BF16(a, b, c) \
  __builtin_amdgcn_wmma_f32_16x16x32_bf16(false, (a), false, (b), (short)0, (c), false, false)

// ---------------------------------------------------------------------------
// Kernel 1: inv[b][q] = 1 / max(sqrt(sum_c x^2), EPS)
// ---------------------------------------------------------------------------
__global__ __launch_bounds__(256) void inv_norm_kernel(const float* __restrict__ x,
                                                       float* __restrict__ inv) {
  int idx = blockIdx.x * blockDim.x + threadIdx.x;   // b*HW + q
  int b = idx >> 10;
  int q = idx & (HW_ - 1);
  const float* xp = x + (size_t)b * C_ * HW_ + q;
  float s = 0.f;
  #pragma unroll 4
  for (int c = 0; c < C_; ++c) {
    float v = xp[(size_t)c * HW_];
    s += v * v;
  }
  inv[idx] = 1.0f / fmaxf(sqrtf(s), EPS_);
}

// ---------------------------------------------------------------------------
// Kernel 2: t[b][o][q] = sum_c W[o][c]*x[b][c][q] + bias[o]   (bf16 output)
// Block: 64(o) x 64(q) tile, 256 threads = 8 waves, each wave 1x2 16x16 tiles.
// ---------------------------------------------------------------------------
__global__ __launch_bounds__(256) void tmat_kernel(const float* __restrict__ x,
                                                   const float* __restrict__ w,
                                                   const float* __restrict__ bias,
                                                   __bf16* __restrict__ t) {
  const int b   = blockIdx.z;
  const int o0  = blockIdx.y * 64;
  const int q0  = blockIdx.x * 64;
  const int tid = threadIdx.x;
  const int lane = tid & 31;
  const int wv   = tid >> 5;
  const int tm   = wv & 3;           // M (o) tile 0..3
  const int tn0  = (wv >> 2) * 2;    // first N (q) tile: 0 or 2

  __shared__ __align__(16) __bf16 sA[64 * 32];    // W tile   [o][c]
  __shared__ __align__(16) __bf16 sBT[64 * 32];   // x tile^T [q][c]

  const float* xb = x + (size_t)b * C_ * HW_;

  f32x8 acc0 = {};
  f32x8 acc1 = {};

  for (int c0 = 0; c0 < C_; c0 += 32) {
    {   // stage A: 64x32, contiguous global reads
      int row = tid >> 2;           // 0..63
      int col = (tid & 3) * 8;      // 0,8,16,24
      const float* src = w + (size_t)(o0 + row) * C_ + c0 + col;
      __bf16* dst = sA + row * 32 + col;
      #pragma unroll
      for (int i = 0; i < 8; ++i) dst[i] = (__bf16)src[i];
    }
    {   // stage B^T: x[c][q] -> sBT[q][c]
      int cl = tid >> 3;            // 0..31
      int qb = (tid & 7) * 8;       // 0..56
      const float* src = xb + (size_t)(c0 + cl) * HW_ + q0 + qb;
      #pragma unroll
      for (int i = 0; i < 8; ++i) sBT[(qb + i) * 32 + cl] = (__bf16)src[i];
    }
    __syncthreads();
    bf16x16 a  = load_a_frag(sA + tm * 16 * 32, 32, lane);
    bf16x16 b0 = load_b_frag(sBT + tn0 * 16 * 32, 32, lane);
    bf16x16 b1 = load_b_frag(sBT + (tn0 + 1) * 16 * 32, 32, lane);
    acc0 = WMMA_BF16(a, b0, acc0);
    acc1 = WMMA_BF16(a, b1, acc1);
    __syncthreads();
  }

  // epilogue: + bias, convert to bf16, store
  const int n  = lane & 15;
  const int r0 = (lane < 16) ? 0 : 8;
  __bf16* tb = t + (size_t)b * C_ * HW_;
  #pragma unroll
  for (int j = 0; j < 8; ++j) {
    int o = o0 + tm * 16 + r0 + j;
    float bv = bias[o];
    tb[(size_t)o * HW_ + q0 + tn0 * 16 + n]       = (__bf16)(acc0[j] + bv);
    tb[(size_t)o * HW_ + q0 + (tn0 + 1) * 16 + n] = (__bf16)(acc1[j] + bv);
  }
}

// ---------------------------------------------------------------------------
// Kernel 3 (fused): out[b][c][q] = sum_k relu(xn_q . xn_k)^2 * t[b][c][k]
// sim never hits memory: per 64-k chunk, compute S[64q][64k] in LDS (stage 1),
// then out-acc += t[64c][64k] @ S^T (stage 2). S in [q][k] layout IS the B^T
// layout needed by stage 2; t[c][k] in global IS the A layout (direct frags).
// ---------------------------------------------------------------------------
__global__ __launch_bounds__(256) void fused_kernel(const float* __restrict__ x,
                                                    const __bf16* __restrict__ t,
                                                    const float* __restrict__ inv,
                                                    float* __restrict__ out) {
  const int b   = blockIdx.z;
  const int co  = blockIdx.y * 64;   // c block (out M)
  const int q0  = blockIdx.x * 64;   // q block (out N)
  const int tid = threadIdx.x;
  const int lane = tid & 31;
  const int wv   = tid >> 5;
  const int tm   = wv & 3;
  const int tn0  = (wv >> 2) * 2;
  const int nn   = lane & 15;
  const int r0   = (lane < 16) ? 0 : 8;

  __shared__ __align__(16) __bf16 sXQ[64 * 32];   // x^T tile [q][c]
  __shared__ __align__(16) __bf16 sXK[64 * 32];   // x^T tile [k][c]
  __shared__ __align__(16) __bf16 sS[64 * 64];    // S tile   [q][k] (== B^T for stage 2)
  __shared__ float sInvQ[64];
  __shared__ float sInvK[64];

  const float*  xb   = x   + (size_t)b * C_ * HW_;
  const __bf16* tb   = t   + (size_t)b * C_ * HW_;
  const float*  invb = inv + b * HW_;

  if (tid < 64) sInvQ[tid] = invb[q0 + tid];

  f32x8 oacc0 = {};
  f32x8 oacc1 = {};

  for (int k0 = 0; k0 < HW_; k0 += 64) {
    if (tid < 64) sInvK[tid] = invb[k0 + tid];

    // ---- stage 1: S = (x[:,q-block]^T x[:,k-chunk]),  M=q N=k K=c ----
    f32x8 sacc0 = {};
    f32x8 sacc1 = {};
    for (int c0 = 0; c0 < C_; c0 += 32) {
      int cl = tid >> 3;
      int qb = (tid & 7) * 8;
      const float* srcq = xb + (size_t)(c0 + cl) * HW_ + q0 + qb;
      const float* srck = xb + (size_t)(c0 + cl) * HW_ + k0 + qb;
      #pragma unroll
      for (int i = 0; i < 8; ++i) {
        sXQ[(qb + i) * 32 + cl] = (__bf16)srcq[i];
        sXK[(qb + i) * 32 + cl] = (__bf16)srck[i];
      }
      __syncthreads();
      bf16x16 a  = load_a_frag(sXQ + tm * 16 * 32, 32, lane);
      bf16x16 b0 = load_b_frag(sXK + tn0 * 16 * 32, 32, lane);
      bf16x16 b1 = load_b_frag(sXK + (tn0 + 1) * 16 * 32, 32, lane);
      sacc0 = WMMA_BF16(a, b0, sacc0);
      sacc1 = WMMA_BF16(a, b1, sacc1);
      __syncthreads();
    }

    // ---- epilogue of stage 1: scale by inv norms, relu, ^2, -> LDS bf16 ----
    #pragma unroll
    for (int j = 0; j < 8; ++j) {
      int ql  = tm * 16 + r0 + j;
      float iq = sInvQ[ql];
      int kl0 = tn0 * 16 + nn;
      int kl1 = kl0 + 16;
      float s0 = fmaxf(sacc0[j] * iq * sInvK[kl0], 0.f);
      float s1 = fmaxf(sacc1[j] * iq * sInvK[kl1], 0.f);
      sS[ql * 64 + kl0] = (__bf16)(s0 * s0);
      sS[ql * 64 + kl1] = (__bf16)(s1 * s1);
    }
    __syncthreads();

    // ---- stage 2: out += t[co:co+64, k-chunk] @ S^T,  M=c N=q K=k ----
    #pragma unroll
    for (int ks = 0; ks < 2; ++ks) {
      bf16x16 a  = load_a_frag(tb + (size_t)(co + tm * 16) * HW_ + k0 + ks * 32,
                               HW_, lane);
      bf16x16 b0 = load_b_frag(sS + tn0 * 16 * 64 + ks * 32, 64, lane);
      bf16x16 b1 = load_b_frag(sS + (tn0 + 1) * 16 * 64 + ks * 32, 64, lane);
      oacc0 = WMMA_BF16(a, b0, oacc0);
      oacc1 = WMMA_BF16(a, b1, oacc1);
    }
    __syncthreads();   // protect sS / sInvK before next chunk
  }

  float* ob = out + (size_t)b * C_ * HW_;
  #pragma unroll
  for (int j = 0; j < 8; ++j) {
    int c = co + tm * 16 + r0 + j;
    ob[(size_t)c * HW_ + q0 + tn0 * 16 + nn]       = oacc0[j];
    ob[(size_t)c * HW_ + q0 + (tn0 + 1) * 16 + nn] = oacc1[j];
  }
}

// ---------------------------------------------------------------------------
extern "C" void kernel_launch(void* const* d_in, const int* in_sizes, int n_in,
                              void* d_out, int out_size, void* d_ws, size_t ws_size,
                              hipStream_t stream) {
  (void)in_sizes; (void)n_in; (void)out_size; (void)ws_size;
  const float* x      = (const float*)d_in[0];
  const float* conv_w = (const float*)d_in[1];
  const float* conv_b = (const float*)d_in[2];
  float* out = (float*)d_out;

  // workspace: [ t (bf16, B*C*HW) | inv (f32, B*HW) ]  ~16.1 MB
  __bf16* tws = (__bf16*)d_ws;
  float*  inv = (float*)((char*)d_ws + (size_t)B_ * C_ * HW_ * sizeof(__bf16));

  inv_norm_kernel<<<dim3((B_ * HW_) / 256), dim3(256), 0, stream>>>(x, inv);
  tmat_kernel<<<dim3(HW_ / 64, C_ / 64, B_), dim3(256), 0, stream>>>(x, conv_w, conv_b, tws);
  fused_kernel<<<dim3(HW_ / 64, C_ / 64, B_), dim3(256), 0, stream>>>(x, tws, inv, out);
}
